// CNN_NCDE_Model_5918464934218
// MI455X (gfx1250) — matrix operand order repaired and checked by reading, not compiled, not run on hardware
//
#include <hip/hip_runtime.h>
#include <hip/hip_bf16.h>

typedef __attribute__((ext_vector_type(16))) __bf16 v16bf;
typedef __attribute__((ext_vector_type(8)))  __bf16 v8bf;
typedef __attribute__((ext_vector_type(8)))  float  v8f;

#define NB   64
#define NCH  32
#define IH   32
#define IW   128
#define PH   16
#define PW   64
#define LSEQ 64
#define CIN  512
#define HID  64
#define F1O  128
#define NSEG 63   // LSEQ-1 spline segments

// Branch-free fast tanh: keeps EXEC uniform (no saveexec around the WMMA loop).
// tanh(x) = 1 - 2/(exp(2x)+1); saturates to +/-1, propagates NaN.
__device__ __forceinline__ float tanh_fast(float x) {
  float ex = __expf(2.f * x);
  return 1.f - 2.f * __builtin_amdgcn_rcpf(ex + 1.f);
}

// ---------------------------------------------------------------- utilities
__global__ void k_f2w_bf16(const float* __restrict__ w, __bf16* __restrict__ o, int n) {
  int i = blockIdx.x * blockDim.x + threadIdx.x;
  if (i < n) o[i] = (__bf16)w[i];
}

// ---------------------------------------------------------------- conv1 5x5 pad2 + relu
__global__ void k_conv1(const float* __restrict__ x, const float* __restrict__ w,
                        const float* __restrict__ b, float* __restrict__ h1) {
  int idx = blockIdx.x * blockDim.x + threadIdx.x;
  if (idx >= NB * NCH * IH * IW) return;
  int xx = idx % IW;
  int y  = (idx / IW) % IH;
  int oc = (idx / (IW * IH)) % NCH;
  int bb = idx / (IW * IH * NCH);
  float s = b[oc];
  for (int ky = 0; ky < 5; ++ky) {
    int iy = y + ky - 2;
    if (iy < 0 || iy >= IH) continue;
    for (int kx = 0; kx < 5; ++kx) {
      int ix = xx + kx - 2;
      if (ix < 0 || ix >= IW) continue;
      s += x[(bb * IH + iy) * IW + ix] * w[oc * 25 + ky * 5 + kx];
    }
  }
  h1[idx] = fmaxf(s, 0.f);
}

// ---------------------------------------------------------------- conv2 3x3 pad1 + relu
__global__ void k_conv2(const float* __restrict__ h1, const float* __restrict__ w,
                        const float* __restrict__ b, float* __restrict__ h2) {
  int idx = blockIdx.x * blockDim.x + threadIdx.x;
  if (idx >= NB * NCH * IH * IW) return;
  int xx = idx % IW;
  int y  = (idx / IW) % IH;
  int oc = (idx / (IW * IH)) % NCH;
  int bb = idx / (IW * IH * NCH);
  float s = b[oc];
  for (int ic = 0; ic < NCH; ++ic) {
    const float* hp = h1 + ((size_t)(bb * NCH + ic) * IH) * IW;
    const float* wp = w + ((oc * NCH + ic) * 9);
    for (int ky = 0; ky < 3; ++ky) {
      int iy = y + ky - 1;
      if (iy < 0 || iy >= IH) continue;
      for (int kx = 0; kx < 3; ++kx) {
        int ix = xx + kx - 1;
        if (ix < 0 || ix >= IW) continue;
        s += hp[iy * IW + ix] * wp[ky * 3 + kx];
      }
    }
  }
  h2[idx] = fmaxf(s, 0.f);
}

// ---------------------------------------------------------------- maxpool 2x2
__global__ void k_pool(const float* __restrict__ h2, float* __restrict__ pool) {
  int idx = blockIdx.x * blockDim.x + threadIdx.x;
  if (idx >= NB * NCH * PH * PW) return;
  int pw = idx % PW;
  int ph = (idx / PW) % PH;
  int c  = (idx / (PW * PH)) % NCH;
  int bb = idx / (PW * PH * NCH);
  const float* p = h2 + (((size_t)(bb * NCH + c) * IH) + 2 * ph) * IW + 2 * pw;
  float m = fmaxf(fmaxf(p[0], p[1]), fmaxf(p[IW], p[IW + 1]));
  pool[idx] = m;
}

// ---------------------------------------------------------------- channel mean
__global__ void k_attmean(const float* __restrict__ pool, float* __restrict__ attin) {
  int idx = blockIdx.x * blockDim.x + threadIdx.x;
  if (idx >= NB * NCH) return;
  const float* p = pool + (size_t)idx * PH * PW;
  float s = 0.f;
  for (int i = 0; i < PH * PW; ++i) s += p[i];
  attin[idx] = s * (1.f / (PH * PW));
}

// ---------------------------------------------------------------- attention MLP
__global__ void k_attmlp(const float* __restrict__ attin,
                         const float* __restrict__ w1, const float* __restrict__ b1,
                         const float* __restrict__ w2, const float* __restrict__ b2,
                         float* __restrict__ att) {
  int idx = blockIdx.x * blockDim.x + threadIdx.x;
  if (idx >= NB * NCH) return;
  int oc = idx % NCH;
  int bb = idx / NCH;
  float hid[4];
  for (int j = 0; j < 4; ++j) {
    float s = b1[j];
    for (int ch = 0; ch < NCH; ++ch) s += attin[bb * NCH + ch] * w1[j * NCH + ch];
    hid[j] = fmaxf(s, 0.f);
  }
  float s = b2[oc];
  for (int j = 0; j < 4; ++j) s += hid[j] * w2[oc * 4 + j];
  att[idx] = 1.f / (1.f + __expf(-s));
}

// ---------------------------------------------------------------- scale + transpose -> seq (B,L,CIN)
__global__ void k_seq(const float* __restrict__ pool, const float* __restrict__ att,
                      float* __restrict__ seq) {
  int idx = blockIdx.x * blockDim.x + threadIdx.x;
  if (idx >= NB * LSEQ * CIN) return;
  int c  = idx % CIN;
  int l  = (idx / CIN) % LSEQ;
  int bb = idx / (CIN * LSEQ);
  int ch = c >> 4;
  int hh = c & 15;
  seq[idx] = pool[((size_t)(bb * NCH + ch) * PH + hh) * PW + l] * att[bb * NCH + ch];
}

// ---------------------------------------------------------------- natural cubic spline (Thomas solve)
__global__ void k_spline(const float* __restrict__ seq, float* __restrict__ sb,
                         float* __restrict__ sc, float* __restrict__ sd) {
  int idx = blockIdx.x * blockDim.x + threadIdx.x;
  if (idx >= NB * CIN) return;
  int c  = idx % CIN;
  int bb = idx / CIN;
  const float* xp = seq + (size_t)bb * LSEQ * CIN + c;
  float xv[LSEQ];
  for (int l = 0; l < LSEQ; ++l) xv[l] = xp[(size_t)l * CIN];
  // solve A m = rhs, A tridiag(1,4,1), n = LSEQ-2 = 62
  float cp[LSEQ - 2], dp[LSEQ - 2];
  cp[0] = 0.25f;
  dp[0] = 6.f * (xv[2] - 2.f * xv[1] + xv[0]) * 0.25f;
  for (int i = 1; i < LSEQ - 2; ++i) {
    float m = 1.f / (4.f - cp[i - 1]);
    cp[i] = m;
    dp[i] = (6.f * (xv[i + 2] - 2.f * xv[i + 1] + xv[i]) - dp[i - 1]) * m;
  }
  float M[LSEQ];
  M[0] = 0.f; M[LSEQ - 1] = 0.f;
  M[LSEQ - 2] = dp[LSEQ - 3];
  for (int i = LSEQ - 4; i >= 0; --i) M[i + 1] = dp[i] - cp[i] * M[i + 2];
  for (int i = 0; i < NSEG; ++i) {
    size_t o = ((size_t)bb * NSEG + i) * CIN + c;
    sb[o] = xv[i + 1] - xv[i] - (2.f * M[i] + M[i + 1]) * (1.f / 6.f);
    sc[o] = M[i] * 0.5f;
    sd[o] = (M[i + 1] - M[i]) * (1.f / 6.f);
  }
}

// ---------------------------------------------------------------- z0 = seq[:,0] @ initial_w^T + b
__global__ void k_z0(const float* __restrict__ seq, const float* __restrict__ iw,
                     const float* __restrict__ ib, float* __restrict__ z) {
  int idx = blockIdx.x * blockDim.x + threadIdx.x;
  if (idx >= NB * HID) return;
  int h  = idx % HID;
  int bb = idx / HID;
  float s = ib[h];
  const float* sp = seq + (size_t)bb * LSEQ * CIN;
  for (int c = 0; c < CIN; ++c) s += sp[c] * iw[h * CIN + c];
  z[idx] = s;
}

// ---------------------------------------------------------------- ztmp = z + a * sum_cg kprev
__global__ void k_zcomb(const float* __restrict__ z, const float* __restrict__ kp,
                        float a, float* __restrict__ ztmp) {
  int i = blockIdx.x * blockDim.x + threadIdx.x;
  if (i >= NB * HID) return;
  float k = kp[i] + kp[4096 + i] + kp[2 * 4096 + i] + kp[3 * 4096 + i];
  ztmp[i] = z[i] + a * k;
}

// ---------------------------------------------------------------- u = relu(ztmp @ f1^T + b) -> bf16
__global__ void k_u(const float* __restrict__ zin, const float* __restrict__ f1w,
                    const float* __restrict__ f1b, __bf16* __restrict__ u) {
  int idx = blockIdx.x * blockDim.x + threadIdx.x;
  if (idx >= NB * F1O) return;
  int i  = idx % F1O;
  int bb = idx / F1O;
  float s = f1b[i];
  for (int k = 0; k < HID; ++k) s += zin[bb * HID + k] * f1w[i * HID + k];
  u[idx] = (__bf16)fmaxf(s, 0.f);
}

// ---------------------------------------------------------------- dXdt(t)
__global__ void k_dx(const float* __restrict__ sb, const float* __restrict__ sc,
                     const float* __restrict__ sd, float t, float* __restrict__ dX) {
  int idx = blockIdx.x * blockDim.x + threadIdx.x;
  if (idx >= NB * CIN) return;
  int c  = idx % CIN;
  int bb = idx / CIN;
  int seg = (int)floorf(t);
  seg = seg < 0 ? 0 : (seg > LSEQ - 2 ? LSEQ - 2 : seg);
  float fr = t - (float)seg;
  size_t o = ((size_t)bb * NSEG + seg) * CIN + c;
  dX[idx] = sb[o] + 2.f * sc[o] * fr + 3.f * sd[o] * fr * fr;
}

// ---------------------------------------------------------------- fused vf: WMMA GEMM + tanh + einsum
// grid (64 h, 4 cg), block 128 (4 waves, one 16-batch tile each)
__global__ __launch_bounds__(128) void k_vf(const __bf16* __restrict__ u,
                                            const __bf16* __restrict__ f2w,
                                            const float*  __restrict__ f2b,
                                            const float*  __restrict__ dX,
                                            float*        __restrict__ kpart) {
  const int h     = blockIdx.x;
  const int cg    = blockIdx.y;
  const int wave  = threadIdx.x >> 5;
  const int lane  = threadIdx.x & 31;
  const int khalf = lane >> 4;
  const int nl    = lane & 15;

  __shared__ __bf16 uls[NB * F1O];           // 16 KB
  for (int i = threadIdx.x; i < NB * F1O; i += 128) uls[i] = u[i];
  __syncthreads();

  // A tile (u, 16x128 bf16): lane nl = batch row, khalf selects K sub-halves
  const int brow = wave * 16 + nl;
  v16bf a[4];
#pragma unroll
  for (int kk = 0; kk < 4; ++kk) {
    const __bf16* p = &uls[brow * F1O + kk * 32 + khalf * 16];
    v8bf lo = *(const v8bf*)p;
    v8bf hi = *(const v8bf*)(p + 8);
#pragma unroll
    for (int e = 0; e < 8; ++e) { a[kk][e] = lo[e]; a[kk][8 + e] = hi[e]; }
  }

  float pacc[8];
#pragma unroll
  for (int r = 0; r < 8; ++r) pacc[r] = 0.f;

  const int base_b = wave * 16 + khalf * 8;

#pragma unroll
  for (int ct = 0; ct < 8; ++ct) {
    const int cglob = cg * 128 + ct * 16 + nl;      // column in CIN, also N = nl
    const int j = h * CIN + cglob;                  // f2_w row
    if (ct < 7) {
      // cover L2 latency of the next c-tile's weight rows (global_prefetch_b8)
      __builtin_prefetch(&f2w[(size_t)(j + 16) * F1O], 0, 3);
    }
    v8f acc = {0.f, 0.f, 0.f, 0.f, 0.f, 0.f, 0.f, 0.f};
#pragma unroll
    for (int kk = 0; kk < 4; ++kk) {
      // B tile (32x16): lane nl = N, K = khalf*16 + e -> 16 contiguous bf16 (32B aligned)
      v16bf bv = *(const v16bf*)&f2w[(size_t)j * F1O + kk * 32 + khalf * 16];
      acc = __builtin_amdgcn_wmma_f32_16x16x32_bf16(false, a[kk], false, bv,
                                                    (short)0, acc, false, false);
    }
    const float bias = f2b[j];
#pragma unroll
    for (int r = 0; r < 8; ++r) {
      float tv = tanh_fast(acc[r] + bias);          // C layout: M = r + 8*khalf, N = nl
      pacc[r] += tv * dX[(size_t)(base_b + r) * CIN + cglob];
    }
  }

  // reduce over the 16 lanes of each half (same batch rows, different c)
#pragma unroll
  for (int m = 1; m < 16; m <<= 1)
#pragma unroll
    for (int r = 0; r < 8; ++r) pacc[r] += __shfl_xor(pacc[r], m, 32);

  if (nl == 0) {
#pragma unroll
    for (int r = 0; r < 8; ++r)
      kpart[(size_t)cg * 4096 + (size_t)(base_b + r) * HID + h] = pacc[r];
  }
}

// ---------------------------------------------------------------- RK4 combine (sums cg partials)
__global__ void k_rk4(float* __restrict__ z, const float* __restrict__ kp) {
  int i = blockIdx.x * blockDim.x + threadIdx.x;
  if (i >= NB * HID) return;
  float k1 = 0, k2 = 0, k3 = 0, k4 = 0;
  for (int cg = 0; cg < 4; ++cg) {
    k1 += kp[(0 * 4 + cg) * 4096 + i];
    k2 += kp[(1 * 4 + cg) * 4096 + i];
    k3 += kp[(2 * 4 + cg) * 4096 + i];
    k4 += kp[(3 * 4 + cg) * 4096 + i];
  }
  z[i] += (0.5f / 6.f) * (k1 + 2.f * k2 + 2.f * k3 + k4);
}

// ---------------------------------------------------------------- final linear (64,2)
__global__ void k_out(const float* __restrict__ z, const float* __restrict__ ow,
                      const float* __restrict__ ob, float* __restrict__ out) {
  int idx = blockIdx.x * blockDim.x + threadIdx.x;
  if (idx >= NB * 2) return;
  int o  = idx & 1;
  int bb = idx >> 1;
  float s = ob[o];
  for (int hh = 0; hh < HID; ++hh) s += z[bb * HID + hh] * ow[o * HID + hh];
  out[idx] = s;
}

// ================================================================ host
extern "C" void kernel_launch(void* const* d_in, const int* in_sizes, int n_in,
                              void* d_out, int out_size, void* d_ws, size_t ws_size,
                              hipStream_t stream) {
  (void)in_sizes; (void)n_in; (void)out_size; (void)ws_size;
  const float* x   = (const float*)d_in[0];
  const float* c1w = (const float*)d_in[1];
  const float* c1b = (const float*)d_in[2];
  const float* c2w = (const float*)d_in[3];
  const float* c2b = (const float*)d_in[4];
  const float* a1w = (const float*)d_in[5];
  const float* a1b = (const float*)d_in[6];
  const float* a2w = (const float*)d_in[7];
  const float* a2b = (const float*)d_in[8];
  const float* iw  = (const float*)d_in[9];
  const float* ib  = (const float*)d_in[10];
  const float* f1w = (const float*)d_in[11];
  const float* f1b = (const float*)d_in[12];
  const float* f2w = (const float*)d_in[13];
  const float* f2b = (const float*)d_in[14];
  const float* ow  = (const float*)d_in[15];
  const float* ob  = (const float*)d_in[16];

  char* ws = (char*)d_ws;
  const size_t sz_h    = (size_t)NB * NCH * IH * IW * 4;   // 33.5 MB
  const size_t sz_pool = (size_t)NB * NCH * PH * PW * 4;   // 8.4 MB
  const size_t sz_seq  = (size_t)NB * LSEQ * CIN * 4;      // 8.4 MB
  const size_t sz_sp   = (size_t)NB * NSEG * CIN * 4;      // 8.26 MB

  size_t off = 0;
  float* h2     = (float*)(ws + off); off += sz_h;
  float* pool   = (float*)(ws + off); off += sz_pool;
  float* attin  = (float*)(ws + off); off += NB * NCH * 4;
  float* att    = (float*)(ws + off); off += NB * NCH * 4;
  float* z      = (float*)(ws + off); off += NB * HID * 4;
  float* ztmp   = (float*)(ws + off); off += NB * HID * 4;
  __bf16* u     = (__bf16*)(ws + off); off += NB * F1O * 2;
  float* dX     = (float*)(ws + off); off += NB * CIN * 4;
  float* kpart  = (float*)(ws + off); off += 4 * 4 * 4096 * 4; // 4 evals x 4 cg x (B*HID)
  __bf16* f2wbf = (__bf16*)(ws + off); off += (size_t)HID * CIN * F1O * 2;
  // union region: h1 lives only between conv1 and conv2; seq/spline reuse it
  float* h1   = (float*)(ws + off);
  float* seq  = (float*)(ws + off);
  float* sb   = (float*)(ws + off + sz_seq);
  float* sc   = (float*)(ws + off + sz_seq + sz_sp);
  float* sd   = (float*)(ws + off + sz_seq + 2 * sz_sp);

  const int T = 256;
  // weight conversion (L2-resident bf16 weights for the WMMA loop)
  k_f2w_bf16<<<(HID * CIN * F1O + T - 1) / T, T, 0, stream>>>(f2w, f2wbf, HID * CIN * F1O);
  // CNN front-end
  k_conv1<<<(NB * NCH * IH * IW + T - 1) / T, T, 0, stream>>>(x, c1w, c1b, h1);
  k_conv2<<<(NB * NCH * IH * IW + T - 1) / T, T, 0, stream>>>(h1, c2w, c2b, h2);
  k_pool<<<(NB * NCH * PH * PW + T - 1) / T, T, 0, stream>>>(h2, pool);
  k_attmean<<<(NB * NCH + T - 1) / T, T, 0, stream>>>(pool, attin);
  k_attmlp<<<(NB * NCH + T - 1) / T, T, 0, stream>>>(attin, a1w, a1b, a2w, a2b, att);
  k_seq<<<(NB * LSEQ * CIN + T - 1) / T, T, 0, stream>>>(pool, att, seq);
  // spline coefficients + initial state
  k_spline<<<(NB * CIN + T - 1) / T, T, 0, stream>>>(seq, sb, sc, sd);
  k_z0<<<(NB * HID + T - 1) / T, T, 0, stream>>>(seq, iw, ib, z);

  // RK4 CDE integration: 126 steps x 4 evals
  const float acoef[4] = {0.f, 0.25f, 0.25f, 0.5f};
  const float toff[4]  = {0.f, 0.25f, 0.25f, 0.5f};
  for (int s = 0; s < 2 * (LSEQ - 1); ++s) {
    float t0 = 0.5f * (float)s;
    for (int e = 0; e < 4; ++e) {
      const float* zin = z;
      if (e > 0) {
        k_zcomb<<<(NB * HID + T - 1) / T, T, 0, stream>>>(z, kpart + (size_t)(e - 1) * 4 * 4096,
                                                          acoef[e], ztmp);
        zin = ztmp;
      }
      k_u<<<(NB * F1O + T - 1) / T, T, 0, stream>>>(zin, f1w, f1b, u);
      if (e != 2)  // e==2 reuses e==1's t
        k_dx<<<(NB * CIN + T - 1) / T, T, 0, stream>>>(sb, sc, sd, t0 + toff[e], dX);
      k_vf<<<dim3(HID, 4), 128, 0, stream>>>(u, f2wbf, f2b, dX, kpart + (size_t)e * 4 * 4096);
    }
    k_rk4<<<(NB * HID + T - 1) / T, T, 0, stream>>>(z, kpart);
  }
  k_out<<<1, 128, 0, stream>>>(z, ow, ob, (float*)d_out);
}
